// HGP_SL_layer_53807350284432
// MI455X (gfx1250) — matrix operand order repaired and checked by reading, not compiled, not run on hardware
//
#include <hip/hip_runtime.h>
#include <hip/hip_bf16.h>

// ---------------- constants from the reference ----------------
#define NNODES 32768
#define NGRAPH 64
#define NPG    512
#define NHEAD  4
#define FHD    64
#define CDIM   256     // NHEAD*FHD
#define FIN    128

typedef __attribute__((ext_vector_type(16))) __bf16 v16bf;
typedef __attribute__((ext_vector_type(8)))  __bf16 v8bf;
typedef __attribute__((ext_vector_type(8)))  float  v8f;

__device__ __forceinline__ unsigned short f2bf(float f) {
    unsigned u = __float_as_uint(f);
    u += 0x7FFFu + ((u >> 16) & 1u);       // round-to-nearest-even
    return (unsigned short)(u >> 16);
}
__device__ __forceinline__ float lrelu02(float v) { return v > 0.f ? v : 0.2f * v; }

__device__ __forceinline__ void atomicMaxF(float* addr, float value) {
    // sign-aware int-punned float max (safe for mixed signs)
    if (value >= 0.f) atomicMax((int*)addr, __float_as_int(value));
    else              atomicMin((unsigned int*)addr, __float_as_uint(value));
}

// ---------------- elementwise converts ----------------
__global__ void k_f32_to_bf16(const float* __restrict__ in, unsigned short* __restrict__ out, int n) {
    int i = blockIdx.x * blockDim.x + threadIdx.x;
    if (i < n) out[i] = f2bf(in[i]);
}

// W [K,Nc] f32 -> Wt [Nc,K] bf16
__global__ void k_transpose_bf16(const float* __restrict__ W, unsigned short* __restrict__ Wt,
                                 int K, int Nc) {
    int i = blockIdx.x * blockDim.x + threadIdx.x;
    if (i >= K * Nc) return;
    int k = i / Nc, n = i % Nc;
    Wt[n * K + k] = f2bf(W[i]);
}

// ---------------- WMMA bf16 GEMM: D = A[M,K] * B[K,Nc] (+bias) ----------------
// A row-major bf16, Bt = B^T row-major bf16 ([Nc,K]); one wave => 16x64 slab.
__global__ void k_wmma_gemm(const unsigned short* __restrict__ Abits,
                            const unsigned short* __restrict__ Btbits,
                            const float* __restrict__ bias,
                            float* __restrict__ D,
                            int M, int K, int Nc) {
    const __bf16* A  = reinterpret_cast<const __bf16*>(Abits);
    const __bf16* Bt = reinterpret_cast<const __bf16*>(Btbits);

    const int lane = threadIdx.x & 31;
    const int wave = (blockIdx.x * blockDim.x + threadIdx.x) >> 5;
    const int nQuads = Nc >> 6;
    const int mt = wave / nQuads;
    const int nq = wave % nQuads;
    if (mt * 16 >= M) return;

    const int ncol = lane & 15;
    const int kb   = (lane >> 4) * 8;    // unused: A loads with kb2, same k order as B
    const int kb2  = (lane >> 4) * 16;   // B k sub-offset (halves interleave at 16)

    const __bf16* Arow = A + (long)(mt * 16 + ncol) * K;
    const __bf16* B0 = Bt + (long)(nq * 64 +  0 + ncol) * K;
    const __bf16* B1 = Bt + (long)(nq * 64 + 16 + ncol) * K;
    const __bf16* B2 = Bt + (long)(nq * 64 + 32 + ncol) * K;
    const __bf16* B3 = Bt + (long)(nq * 64 + 48 + ncol) * K;

    v8f acc0 = {}, acc1 = {}, acc2 = {}, acc3 = {};

    for (int k0 = 0; k0 < K; k0 += 32) {
        v16bf a = *(const v16bf*)(Arow + k0 + kb2);
        v16bf b0 = *(const v16bf*)(B0 + k0 + kb2);
        v16bf b1 = *(const v16bf*)(B1 + k0 + kb2);
        v16bf b2 = *(const v16bf*)(B2 + k0 + kb2);
        v16bf b3 = *(const v16bf*)(B3 + k0 + kb2);
        acc0 = __builtin_amdgcn_wmma_f32_16x16x32_bf16(false, a, false, b0, (short)0, acc0, false, false);
        acc1 = __builtin_amdgcn_wmma_f32_16x16x32_bf16(false, a, false, b1, (short)0, acc1, false, false);
        acc2 = __builtin_amdgcn_wmma_f32_16x16x32_bf16(false, a, false, b2, (short)0, acc2, false, false);
        acc3 = __builtin_amdgcn_wmma_f32_16x16x32_bf16(false, a, false, b3, (short)0, acc3, false, false);
    }

    float bv0 = 0.f, bv1 = 0.f, bv2 = 0.f, bv3 = 0.f;
    if (bias) {
        bv0 = bias[nq * 64 +  0 + ncol];
        bv1 = bias[nq * 64 + 16 + ncol];
        bv2 = bias[nq * 64 + 32 + ncol];
        bv3 = bias[nq * 64 + 48 + ncol];
    }
    const int mbase = mt * 16 + (lane >> 4) * 8;   // C/D layout: VGPR r -> row mbase+r
    #pragma unroll
    for (int r = 0; r < 8; ++r) {
        float* drow = D + (long)(mbase + r) * Nc + nq * 64 + ncol;
        drow[0]  = acc0[r] + bv0;
        drow[16] = acc1[r] + bv1;
        drow[32] = acc2[r] + bv2;
        drow[48] = acc3[r] + bv3;
    }
}

// ---------------- GAT edge softmax / aggregation ----------------
// es/ed per (node, head)
__global__ void k_attn_dots(const float* __restrict__ h, const float* __restrict__ a_src,
                            const float* __restrict__ a_dst,
                            float* __restrict__ es, float* __restrict__ ed, int n) {
    int idx = blockIdx.x * blockDim.x + threadIdx.x;
    if (idx >= n * NHEAD) return;
    int node = idx >> 2, hh = idx & 3;
    const float* hp = h + (long)node * CDIM + hh * FHD;
    const float* as = a_src + hh * FHD;
    const float* ad = a_dst + hh * FHD;
    float s1 = 0.f, s2 = 0.f;
    #pragma unroll 4
    for (int i = 0; i < FHD; ++i) { float hv = hp[i]; s1 += hv * as[i]; s2 += hv * ad[i]; }
    es[idx] = s1; ed[idx] = s2;
}

__global__ void k_init_m(const float* __restrict__ es, const float* __restrict__ ed,
                         float* __restrict__ m, int n4) {
    int i = blockIdx.x * blockDim.x + threadIdx.x;
    if (i < n4) m[i] = lrelu02(es[i] + ed[i]);   // self-loop value
}

__global__ void k_edge_max(const int* __restrict__ src, const int* __restrict__ dst,
                           const float* __restrict__ es, const float* __restrict__ ed,
                           float* __restrict__ m, int E) {
    int e = blockIdx.x * blockDim.x + threadIdx.x;
    if (e >= E) return;
    int s = src[e], d = dst[e];
    #pragma unroll
    for (int hh = 0; hh < NHEAD; ++hh) {
        float v = lrelu02(es[s * NHEAD + hh] + ed[d * NHEAD + hh]);
        atomicMaxF(&m[d * NHEAD + hh], v);
    }
}

__global__ void k_init_den(const float* __restrict__ es, const float* __restrict__ ed,
                           const float* __restrict__ m, float* __restrict__ den, int n4) {
    int i = blockIdx.x * blockDim.x + threadIdx.x;
    if (i < n4) den[i] = __expf(lrelu02(es[i] + ed[i]) - m[i]);   // self-loop term
}

__global__ void k_edge_sum(const int* __restrict__ src, const int* __restrict__ dst,
                           const float* __restrict__ es, const float* __restrict__ ed,
                           const float* __restrict__ m, float* __restrict__ den, int E) {
    int e = blockIdx.x * blockDim.x + threadIdx.x;
    if (e >= E) return;
    int s = src[e], d = dst[e];
    #pragma unroll
    for (int hh = 0; hh < NHEAD; ++hh) {
        float v = lrelu02(es[s * NHEAD + hh] + ed[d * NHEAD + hh]);
        atomicAdd(&den[d * NHEAD + hh], __expf(v - m[d * NHEAD + hh]));
    }
}

// out[n,:] = alpha_self * h[n,:]
__global__ void k_self_init(const float* __restrict__ h, const float* __restrict__ es,
                            const float* __restrict__ ed, const float* __restrict__ m,
                            const float* __restrict__ den, float* __restrict__ out, int n) {
    long i = (long)blockIdx.x * blockDim.x + threadIdx.x;
    if (i >= (long)n * CDIM) return;
    int node = (int)(i >> 8), f = (int)(i & 255), hh = f >> 6;
    int a = node * NHEAD + hh;
    float v = lrelu02(es[a] + ed[a]);
    float alpha = __expf(v - m[a]) / (den[a] + 1e-16f);
    out[i] = alpha * h[i];
}

// wave per edge, 8 floats per lane: out[dst] += alpha * h[src]
__global__ void k_edge_aggregate(const int* __restrict__ src, const int* __restrict__ dst,
                                 const float* __restrict__ h, const float* __restrict__ es,
                                 const float* __restrict__ ed, const float* __restrict__ m,
                                 const float* __restrict__ den, float* __restrict__ out, int E) {
    long gtid = (long)blockIdx.x * blockDim.x + threadIdx.x;
    int e = (int)(gtid >> 5);
    if (e >= E) return;
    int lane = (int)(gtid & 31);
    int s = src[e], d = dst[e];
    int hh = lane >> 3;            // 8 consecutive features live inside one head
    int a = d * NHEAD + hh;
    float v = lrelu02(es[s * NHEAD + hh] + ed[a]);
    float alpha = __expf(v - m[a]) / (den[a] + 1e-16f);
    int f0 = lane * 8;
    const float4* hs = (const float4*)(h + (long)s * CDIM + f0);
    float4 p0 = hs[0], p1 = hs[1];
    float* o = out + (long)d * CDIM + f0;
    atomicAdd(o + 0, alpha * p0.x); atomicAdd(o + 1, alpha * p0.y);
    atomicAdd(o + 2, alpha * p0.z); atomicAdd(o + 3, alpha * p0.w);
    atomicAdd(o + 4, alpha * p1.x); atomicAdd(o + 5, alpha * p1.y);
    atomicAdd(o + 6, alpha * p1.z); atomicAdd(o + 7, alpha * p1.w);
}

// out = elu(out + bias), plus bf16 copy for next WMMA GEMM
__global__ void k_bias_elu_bf16(float* __restrict__ out, const float* __restrict__ bias,
                                unsigned short* __restrict__ cb, int n) {
    long i = (long)blockIdx.x * blockDim.x + threadIdx.x;
    if (i >= (long)n * CDIM) return;
    float v = out[i] + bias[(int)(i & 255)];
    v = v > 0.f ? v : __expf(v) - 1.f;
    out[i] = v;
    cb[i] = f2bf(v);
}

// ---------------- pooling ----------------
__global__ void k_scores(const float* __restrict__ h, const float* __restrict__ sw,
                         const float* __restrict__ sb, float* __restrict__ scores, int M) {
    int n = blockIdx.x * blockDim.x + threadIdx.x;
    if (n >= M) return;
    const float4* hp = (const float4*)(h + (long)n * CDIM);
    const float4* wp = (const float4*)sw;
    float s = 0.f;
    #pragma unroll 4
    for (int i = 0; i < CDIM / 4; ++i) {
        float4 a = hp[i], b = wp[i];
        s += a.x * b.x + a.y * b.y + a.z * b.z + a.w * b.w;
    }
    s += sb[0];
    scores[n] = 1.f / (1.f + __expf(-s));
}

// per-graph top-k by rank counting (matches top_k tie-break: score desc, index asc)
__global__ void k_topk(const float* __restrict__ scores, int* __restrict__ sel,
                       int npg, int k) {
    __shared__ float sc[NPG];
    int g = blockIdx.x, i = threadIdx.x;
    sc[i] = scores[g * npg + i];
    __syncthreads();
    float si = sc[i];
    int rank = 0;
    for (int j = 0; j < npg; ++j) {
        float sj = sc[j];
        rank += (sj > si) || (sj == si && j < i);
    }
    if (rank < k) sel[g * k + rank] = g * npg + i;
}

__global__ void k_gather_bf16(const float* __restrict__ h, const int* __restrict__ sel,
                              unsigned short* __restrict__ out, int rows) {
    long i = (long)blockIdx.x * blockDim.x + threadIdx.x;
    if (i >= (long)rows * CDIM) return;
    int r = (int)(i >> 8), f = (int)(i & 255);
    out[i] = f2bf(h[(long)sel[r] * CDIM + f]);
}

// ---------------- classifier: mean -> 256x64 relu -> 64x2 -> log_softmax ----------------
__global__ void k_classify(const float* __restrict__ pooled,
                           const float* __restrict__ c1w, const float* __restrict__ c1b,
                           const float* __restrict__ c2w, const float* __restrict__ c2b,
                           float* __restrict__ out, int k2) {
    __shared__ float g[CDIM];
    __shared__ float hid[64];
    __shared__ float z[2];
    int gi = blockIdx.x, t = threadIdx.x;
    const float* base = pooled + (long)gi * k2 * CDIM;
    float s = 0.f;
    for (int r = 0; r < k2; ++r) s += base[(long)r * CDIM + t];
    g[t] = s / (float)k2;
    __syncthreads();
    if (t < 64) {
        float h = c1b[t];
        for (int f = 0; f < CDIM; ++f) h += g[f] * c1w[f * 64 + t];
        hid[t] = h > 0.f ? h : 0.f;
    }
    __syncthreads();
    if (t < 2) {
        float zz = c2b[t];
        for (int j = 0; j < 64; ++j) zz += hid[j] * c2w[j * 2 + t];
        z[t] = zz;
    }
    __syncthreads();
    if (t < 2) {
        float mx = fmaxf(z[0], z[1]);
        float lse = mx + __logf(__expf(z[0] - mx) + __expf(z[1] - mx));
        out[gi * 2 + t] = z[t] - lse;
    }
}

// ---------------- launcher ----------------
static inline int cdiv(long a, long b) { return (int)((a + b - 1) / b); }

extern "C" void kernel_launch(void* const* d_in, const int* in_sizes, int n_in,
                              void* d_out, int out_size, void* d_ws, size_t ws_size,
                              hipStream_t stream) {
    const float* x      = (const float*)d_in[0];
    const int*   ei     = (const int*)d_in[1];
    const float* W1     = (const float*)d_in[3];
    const float* a_src1 = (const float*)d_in[4];
    const float* a_dst1 = (const float*)d_in[5];
    const float* b1     = (const float*)d_in[6];
    const float* W2     = (const float*)d_in[7];
    const float* a_src2 = (const float*)d_in[8];
    const float* a_dst2 = (const float*)d_in[9];
    const float* b2     = (const float*)d_in[10];
    const float* p1_sw  = (const float*)d_in[11];
    const float* p1_sb  = (const float*)d_in[12];
    const float* p1_tw  = (const float*)d_in[13];
    const float* p1_tb  = (const float*)d_in[14];
    const float* p2_sw  = (const float*)d_in[15];
    const float* p2_sb  = (const float*)d_in[16];
    const float* p2_tw  = (const float*)d_in[17];
    const float* p2_tb  = (const float*)d_in[18];
    const float* c1w    = (const float*)d_in[19];
    const float* c1b    = (const float*)d_in[20];
    const float* c2w    = (const float*)d_in[21];
    const float* c2b    = (const float*)d_in[22];

    const int E = in_sizes[1] / 2;
    const int* src = ei;
    const int* dst = ei + E;

    const int K1 = NPG / 2;        // 256 kept per graph after pool1
    const int K2 = K1 / 2;         // 128 after pool2
    const int R1 = NGRAPH * K1;    // 16384 rows
    const int R2 = NGRAPH * K2;    // 8192 rows

    // workspace carve (256B aligned)
    char* ws = (char*)d_ws;
    size_t off = 0;
    auto carve = [&](size_t bytes) -> void* {
        off = (off + 255) & ~(size_t)255;
        void* p = ws + off;
        off += bytes;
        return p;
    };
    float*          bufA  = (float*)carve((size_t)NNODES * CDIM * 4);   // lin features / pool1 out
    float*          bufB  = (float*)carve((size_t)NNODES * CDIM * 4);   // aggregated / pool2 out
    unsigned short* cb    = (unsigned short*)carve((size_t)NNODES * CDIM * 2); // bf16 activations
    unsigned short* wt    = (unsigned short*)carve((size_t)CDIM * CDIM * 2);   // bf16 W^T
    float*          es    = (float*)carve((size_t)NNODES * NHEAD * 4);
    float*          ed    = (float*)carve((size_t)NNODES * NHEAD * 4);
    float*          mbuf  = (float*)carve((size_t)NNODES * NHEAD * 4);
    float*          den   = (float*)carve((size_t)NNODES * NHEAD * 4);
    float*          sco   = (float*)carve((size_t)NNODES * 4);
    int*            sel1  = (int*)carve((size_t)R1 * 4);
    int*            sel2  = (int*)carve((size_t)R2 * 4);

    const int T = 256;
    auto gemm_blocks = [&](int M, int Nc) { return ((M / 16) * (Nc / 64) * 32 + T - 1) / T; };

    // ---- GAT layer 1 ----
    k_f32_to_bf16<<<cdiv((long)NNODES * FIN, T), T, 0, stream>>>(x, cb, NNODES * FIN);
    k_transpose_bf16<<<cdiv((long)FIN * CDIM, T), T, 0, stream>>>(W1, wt, FIN, CDIM);
    k_wmma_gemm<<<gemm_blocks(NNODES, CDIM), T, 0, stream>>>(cb, wt, nullptr, bufA, NNODES, FIN, CDIM);
    k_attn_dots<<<cdiv((long)NNODES * NHEAD, T), T, 0, stream>>>(bufA, a_src1, a_dst1, es, ed, NNODES);
    k_init_m<<<cdiv((long)NNODES * NHEAD, T), T, 0, stream>>>(es, ed, mbuf, NNODES * NHEAD);
    k_edge_max<<<cdiv(E, T), T, 0, stream>>>(src, dst, es, ed, mbuf, E);
    k_init_den<<<cdiv((long)NNODES * NHEAD, T), T, 0, stream>>>(es, ed, mbuf, den, NNODES * NHEAD);
    k_edge_sum<<<cdiv(E, T), T, 0, stream>>>(src, dst, es, ed, mbuf, den, E);
    k_self_init<<<cdiv((long)NNODES * CDIM, T), T, 0, stream>>>(bufA, es, ed, mbuf, den, bufB, NNODES);
    k_edge_aggregate<<<cdiv((long)E * 32, T), T, 0, stream>>>(src, dst, bufA, es, ed, mbuf, den, bufB, E);
    k_bias_elu_bf16<<<cdiv((long)NNODES * CDIM, T), T, 0, stream>>>(bufB, b1, cb, NNODES);

    // ---- GAT layer 2 ----
    k_transpose_bf16<<<cdiv((long)CDIM * CDIM, T), T, 0, stream>>>(W2, wt, CDIM, CDIM);
    k_wmma_gemm<<<gemm_blocks(NNODES, CDIM), T, 0, stream>>>(cb, wt, nullptr, bufA, NNODES, CDIM, CDIM);
    k_attn_dots<<<cdiv((long)NNODES * NHEAD, T), T, 0, stream>>>(bufA, a_src2, a_dst2, es, ed, NNODES);
    k_init_m<<<cdiv((long)NNODES * NHEAD, T), T, 0, stream>>>(es, ed, mbuf, NNODES * NHEAD);
    k_edge_max<<<cdiv(E, T), T, 0, stream>>>(src, dst, es, ed, mbuf, E);
    k_init_den<<<cdiv((long)NNODES * NHEAD, T), T, 0, stream>>>(es, ed, mbuf, den, NNODES * NHEAD);
    k_edge_sum<<<cdiv(E, T), T, 0, stream>>>(src, dst, es, ed, mbuf, den, E);
    k_self_init<<<cdiv((long)NNODES * CDIM, T), T, 0, stream>>>(bufA, es, ed, mbuf, den, bufB, NNODES);
    k_edge_aggregate<<<cdiv((long)E * 32, T), T, 0, stream>>>(src, dst, bufA, es, ed, mbuf, den, bufB, E);
    k_bias_elu_bf16<<<cdiv((long)NNODES * CDIM, T), T, 0, stream>>>(bufB, b2, cb, NNODES);
    // bufB = h2 (f32)

    // ---- pool 1: score -> top-256/512 -> transform ----
    k_scores<<<cdiv(NNODES, T), T, 0, stream>>>(bufB, p1_sw, p1_sb, sco, NNODES);
    k_topk<<<NGRAPH, NPG, 0, stream>>>(sco, sel1, NPG, K1);
    k_gather_bf16<<<cdiv((long)R1 * CDIM, T), T, 0, stream>>>(bufB, sel1, cb, R1);
    k_transpose_bf16<<<cdiv((long)CDIM * CDIM, T), T, 0, stream>>>(p1_tw, wt, CDIM, CDIM);
    k_wmma_gemm<<<gemm_blocks(R1, CDIM), T, 0, stream>>>(cb, wt, p1_tb, bufA, R1, CDIM, CDIM);

    // ---- pool 2: score -> top-128/256 -> transform ----
    k_scores<<<cdiv(R1, T), T, 0, stream>>>(bufA, p2_sw, p2_sb, sco, R1);
    k_topk<<<NGRAPH, K1, 0, stream>>>(sco, sel2, K1, K2);
    k_gather_bf16<<<cdiv((long)R2 * CDIM, T), T, 0, stream>>>(bufA, sel2, cb, R2);
    k_transpose_bf16<<<cdiv((long)CDIM * CDIM, T), T, 0, stream>>>(p2_tw, wt, CDIM, CDIM);
    k_wmma_gemm<<<gemm_blocks(R2, CDIM), T, 0, stream>>>(cb, wt, p2_tb, bufB, R2, CDIM, CDIM);

    // ---- mean pool + MLP + log_softmax ----
    k_classify<<<NGRAPH, CDIM, 0, stream>>>(bufB, c1w, c1b, c2w, c2b, (float*)d_out, K2);
}
